// EnhancedLSTMDecoder_88579405512927
// MI455X (gfx1250) — matrix operand order repaired and checked by reading, not compiled, not run on hardware
//
#include <hip/hip_runtime.h>
#include <hip/hip_bf16.h>

// ---------------------------------------------------------------------------
// EnhancedLSTMDecoder for MI455X (gfx1250): bf16 WMMA GEMMs + VALU attention.
// V=30000 E=768 H=768 NH=8 HD=96 L=3 B=128 T=32 S=196
// All GEMM operands pre-converted to bf16 once per launch. K is always 768 ->
// fully unrolled, register double-buffered K loop: loads for step i+1 issue
// before the WMMAs of step i wait, so s_wait_loadcnt is partial, not a drain.
// ---------------------------------------------------------------------------

#define BDIM 256
#define WPB  8   // waves per block (wave32)
#define KDIM 768

typedef __attribute__((ext_vector_type(16))) __bf16 v16bf;
typedef __attribute__((ext_vector_type(8)))  float  v8f;

__device__ __forceinline__ unsigned int f2bf(float f) {
  unsigned int u = __float_as_uint(f);
  return (u + 0x7FFFu + ((u >> 16) & 1u)) >> 16;   // round-to-nearest-even
}
__device__ __forceinline__ float bf2f(unsigned short h) {
  return __uint_as_float(((unsigned int)h) << 16);
}
__device__ __forceinline__ float bf2f_lo(unsigned int w) { return __uint_as_float(w << 16); }
__device__ __forceinline__ float bf2f_hi(unsigned int w) { return __uint_as_float(w & 0xFFFF0000u); }
__device__ __forceinline__ unsigned int packbf(float lo, float hi) {
  return f2bf(lo) | (f2bf(hi) << 16);
}
__device__ __forceinline__ float sigf(float x) { return 1.0f / (1.0f + expf(-x)); }

union FragU { v16bf v; unsigned int u[8]; };

// A fragment (16x32 bf16): lanes 0-15 -> rows M=0..15, K={0..7,16..23};
// lanes 16-31 -> same rows, K={8..15,24..31}. Dword j packs K=2j(lo),2j+1(hi).
__device__ __forceinline__ void loadA(FragU& f, const unsigned short* __restrict__ X,
                                      int ld, int row, int kb, int lane) {
  const unsigned short* p = X + (size_t)row * ld + kb + ((lane >> 4) << 3);
  uint4 a = *reinterpret_cast<const uint4*>(p);
  uint4 b = *reinterpret_cast<const uint4*>(p + 16);
  f.u[0] = a.x; f.u[1] = a.y; f.u[2] = a.z; f.u[3] = a.w;
  f.u[4] = b.x; f.u[5] = b.y; f.u[6] = b.z; f.u[7] = b.w;
}

// B fragment (32x16 bf16): lane -> column N=lane&15; lanes 0-15 K={0..7,16..23},
// lanes 16-31 K={8..15,24..31}. W is (N x K) row-major.
__device__ __forceinline__ void loadB(FragU& f, const unsigned short* __restrict__ W,
                                      int ld, int col, int kb, int lane) {
  const unsigned short* p = W + (size_t)col * ld + kb + ((lane >> 4) << 3);
  uint4 a = *reinterpret_cast<const uint4*>(p);
  uint4 b = *reinterpret_cast<const uint4*>(p + 16);
  f.u[0] = a.x; f.u[1] = a.y; f.u[2] = a.z; f.u[3] = a.w;
  f.u[4] = b.x; f.u[5] = b.y; f.u[6] = b.z; f.u[7] = b.w;
}

__device__ __forceinline__ v8f wmma_bf(const FragU& a, const FragU& b, v8f c) {
  return __builtin_amdgcn_wmma_f32_16x16x32_bf16(false, a.v, false, b.v,
                                                 (short)0, c, false, false);
}

// K=768 fully unrolled, register double-buffered, two independent WMMA chains
__device__ __forceinline__ void kloop_dual(v8f& acc0, v8f& acc1,
                                           const unsigned short* __restrict__ X, int ldx,
                                           const unsigned short* __restrict__ W, int ldw,
                                           int rowA, int colB, int lane) {
  FragU fa[2], fb0[2], fb1[2];
  loadA(fa[0], X, ldx, rowA, 0, lane);
  loadB(fb0[0], W, ldw, colB, 0, lane);
  loadB(fb1[0], W, ldw, colB + 16, 0, lane);
#pragma unroll
  for (int i = 0; i < KDIM / 32; ++i) {
    int cur = i & 1, nxt = cur ^ 1;
    if (i + 1 < KDIM / 32) {
      loadA(fa[nxt], X, ldx, rowA, (i + 1) * 32, lane);
      loadB(fb0[nxt], W, ldw, colB, (i + 1) * 32, lane);
      loadB(fb1[nxt], W, ldw, colB + 16, (i + 1) * 32, lane);
    }
    acc0 = wmma_bf(fa[cur], fb0[cur], acc0);
    acc1 = wmma_bf(fa[cur], fb1[cur], acc1);
  }
}

__device__ __forceinline__ void kloop_single(v8f& acc0,
                                             const unsigned short* __restrict__ X, int ldx,
                                             const unsigned short* __restrict__ W, int ldw,
                                             int rowA, int colB, int lane) {
  FragU fa[2], fb0[2];
  loadA(fa[0], X, ldx, rowA, 0, lane);
  loadB(fb0[0], W, ldw, colB, 0, lane);
#pragma unroll
  for (int i = 0; i < KDIM / 32; ++i) {
    int cur = i & 1, nxt = cur ^ 1;
    if (i + 1 < KDIM / 32) {
      loadA(fa[nxt], X, ldx, rowA, (i + 1) * 32, lane);
      loadB(fb0[nxt], W, ldw, colB, (i + 1) * 32, lane);
    }
    acc0 = wmma_bf(fa[cur], fb0[cur], acc0);
  }
}

// Each wave computes a 16x32 output strip (two tiles) when possible.
// epi: 0 = identity, 2 = sigmoid gate blend with aux1/aux2, 3 = exact GELU.
// K1/K2 are 768 (K2 may be 0). Result -> outf (f32) and/or outh (bf16).
__global__ void k_gemm(const unsigned short* __restrict__ X1, int ldx1,
                       const unsigned short* __restrict__ W1, int ldw1,
                       const unsigned short* __restrict__ X2, int ldx2,
                       const unsigned short* __restrict__ W2, int ldw2, int K2,
                       const float* __restrict__ bias, const float* __restrict__ bias2,
                       int M, int N,
                       float* __restrict__ outf, unsigned short* __restrict__ outh, int ldo,
                       int epi, const float* __restrict__ aux1,
                       const float* __restrict__ aux2, int ldaux) {
  int wave = __builtin_amdgcn_readfirstlane(threadIdx.x >> 5);   // scalar wave id
  int lane = threadIdx.x & 31;
  int ntp = (N + 31) >> 5;                    // N-tile pairs
  long tile = (long)blockIdx.x * WPB + wave;
  long total = (long)(M >> 4) * ntp;
  if (tile >= total) return;                  // scalar -> EXEC stays full
  int mt = (int)(tile / ntp), np = (int)(tile % ntp);
  int m0 = mt << 4, n0 = np << 5;
  bool has2 = (n0 + 16) < N;                  // scalar, hoisted out of K loop

  v8f acc0 = {0.f,0.f,0.f,0.f,0.f,0.f,0.f,0.f};
  v8f acc1 = {0.f,0.f,0.f,0.f,0.f,0.f,0.f,0.f};
  int rowA = m0 + (lane & 15);
  int colB0 = n0 + (lane & 15);
  if (has2) {
    kloop_dual(acc0, acc1, X1, ldx1, W1, ldw1, rowA, colB0, lane);
    if (K2) kloop_dual(acc0, acc1, X2, ldx2, W2, ldw2, rowA, colB0, lane);
  } else {
    kloop_single(acc0, X1, ldx1, W1, ldw1, rowA, colB0, lane);
    if (K2) kloop_single(acc0, X2, ldx2, W2, ldw2, rowA, colB0, lane);
  }

  // C layout: dword r -> row m0 + 8*(lane>=16) + r, col n + (lane&15)
  int rbase = m0 + ((lane >> 4) << 3);
  for (int half = 0; half < (has2 ? 2 : 1); ++half) {
    int col = n0 + (half << 4) + (lane & 15);
    const v8f& acc = half ? acc1 : acc0;
    float bv = bias ? bias[col] : 0.0f;
    if (bias2) bv += bias2[col];
#pragma unroll
    for (int r = 0; r < 8; ++r) {
      int row = rbase + r;
      float v = acc[r] + bv;
      float res;
      if (epi == 2) {
        float s = 1.0f / (1.0f + expf(-v));
        size_t ao = (size_t)row * ldaux + col;
        res = s * aux1[ao] + (1.0f - s) * aux2[ao];
      } else if (epi == 3) {
        res = 0.5f * v * (1.0f + erff(v * 0.70710678118654752f));
      } else {
        res = v;
      }
      size_t o = (size_t)row * ldo + col;
      if (outf) outf[o] = res;
      if (outh) outh[o] = (unsigned short)f2bf(res);
    }
  }
}

// one-time f32 -> bf16 conversion (vectorized x4)
__global__ void k_f2bf(const float* __restrict__ in, unsigned short* __restrict__ out, long n) {
  long i = ((long)blockIdx.x * BDIM + threadIdx.x) * 4;
  if (i + 3 < n) {
    float4 v = *reinterpret_cast<const float4*>(in + i);
    uint2 o;
    o.x = packbf(v.x, v.y);
    o.y = packbf(v.z, v.w);
    *reinterpret_cast<uint2*>(out + i) = o;
  } else {
    for (; i < n; ++i) out[i] = (unsigned short)f2bf(in[i]);
  }
}

// token embedding + positional encoding: (T,B,E) in f32 and bf16
__global__ void k_embed(const int* __restrict__ cap, const float* __restrict__ embW,
                        const float* __restrict__ pos, float* __restrict__ emb,
                        unsigned short* __restrict__ embbf) {
  long idx = (long)blockIdx.x * BDIM + threadIdx.x;
  if (idx >= (long)32 * 128 * 768) return;
  int e = (int)(idx % 768);
  long tb = idx / 768;
  int b = (int)(tb % 128);
  int t = (int)(tb / 128);
  int tok = cap[t * 128 + b];
  float v = embW[(size_t)tok * 768 + e] + pos[(size_t)t * 768 + e];
  emb[idx] = v;
  embbf[idx] = (unsigned short)f2bf(v);
}

// single-query multi-head cross attention over S=196 (K/V bf16, L2-resident)
// one block per batch row; writes ctx_pre (bf16) and head-mean attention (B,S)
__global__ void k_attn(const float* __restrict__ q, const unsigned short* __restrict__ Kb,
                       const unsigned short* __restrict__ Vb,
                       unsigned short* __restrict__ ctxp_bf, float* __restrict__ attn_out) {
  __shared__ float qs[768];
  __shared__ float pAll[8 * 196];
  __shared__ float red[BDIM];
  int b = blockIdx.x, tid = threadIdx.x;
  for (int i = tid; i < 768; i += BDIM) qs[i] = q[(size_t)b * 768 + i];
  __syncthreads();
  const float scale = 0.1020620726159657f;  // 1/sqrt(96)
  for (int h = 0; h < 8; ++h) {
    float sc = -1e30f;
    if (tid < 196) {
      const unsigned int* kr = reinterpret_cast<const unsigned int*>(
          Kb + ((size_t)(b * 196 + tid)) * 768 + h * 96);
      float a = 0.f;
#pragma unroll 4
      for (int d = 0; d < 48; ++d) {
        unsigned int w = kr[d];
        a += qs[h * 96 + 2 * d]     * bf2f_lo(w);
        a += qs[h * 96 + 2 * d + 1] * bf2f_hi(w);
      }
      sc = a * scale;
    }
    red[tid] = sc;
    __syncthreads();
    for (int off = 128; off > 0; off >>= 1) {
      if (tid < off) red[tid] = fmaxf(red[tid], red[tid + off]);
      __syncthreads();
    }
    float mx = red[0];
    __syncthreads();
    float ex = (tid < 196) ? expf(sc - mx) : 0.f;
    red[tid] = ex;
    __syncthreads();
    for (int off = 128; off > 0; off >>= 1) {
      if (tid < off) red[tid] += red[tid + off];
      __syncthreads();
    }
    float inv = 1.0f / red[0];
    __syncthreads();
    if (tid < 196) pAll[h * 196 + tid] = ex * inv;
    __syncthreads();
  }
  for (int e = tid; e < 768; e += BDIM) {
    int h = e / 96;
    const float* p = pAll + h * 196;
    const unsigned short* vr = Vb + (size_t)b * 196 * 768 + e;
    float a = 0.f;
    for (int s = 0; s < 196; ++s) a += p[s] * bf2f(vr[(size_t)s * 768]);
    ctxp_bf[(size_t)b * 768 + e] = (unsigned short)f2bf(a);
  }
  if (tid < 196) {
    float a = 0.f;
    for (int h = 0; h < 8; ++h) a += pAll[h * 196 + tid];
    attn_out[(size_t)b * 196 + tid] = a * 0.125f;
  }
}

// LSTM cell pointwise: gates (B,4H) i,f,g,o -> update c, produce raw h
__global__ void k_lstm_point(const float* __restrict__ gates, float* __restrict__ c,
                             float* __restrict__ hraw) {
  int idx = blockIdx.x * BDIM + threadIdx.x;
  if (idx >= 128 * 768) return;
  int b = idx / 768, j = idx % 768;
  const float* g = gates + (size_t)b * 3072;
  float ig = g[j], fg = g[768 + j], gg = g[1536 + j], og = g[2304 + j];
  float cn = sigf(fg) * c[idx] + sigf(ig) * tanhf(gg);
  c[idx] = cn;
  hraw[idx] = sigf(og) * tanhf(cn);
}

// LayerNorm over H=768 per row; writes f32 + bf16; optional extra f32 copy
__global__ void k_ln(const float* __restrict__ hraw, const float* __restrict__ g,
                     const float* __restrict__ bta, float* __restrict__ hout,
                     unsigned short* __restrict__ houtbf, float* __restrict__ hcopy) {
  __shared__ float red[BDIM];
  int b = blockIdx.x, tid = threadIdx.x;
  const float* row = hraw + (size_t)b * 768;
  float s = 0.f;
  for (int i = tid; i < 768; i += BDIM) s += row[i];
  red[tid] = s; __syncthreads();
  for (int off = 128; off > 0; off >>= 1) { if (tid < off) red[tid] += red[tid + off]; __syncthreads(); }
  float mu = red[0] * (1.0f / 768.0f);
  __syncthreads();
  float v = 0.f;
  for (int i = tid; i < 768; i += BDIM) { float d = row[i] - mu; v += d * d; }
  red[tid] = v; __syncthreads();
  for (int off = 128; off > 0; off >>= 1) { if (tid < off) red[tid] += red[tid + off]; __syncthreads(); }
  float rs = rsqrtf(red[0] * (1.0f / 768.0f) + 1e-5f);
  for (int i = tid; i < 768; i += BDIM) {
    float o = (row[i] - mu) * rs * g[i] + bta[i];
    hout[(size_t)b * 768 + i] = o;
    houtbf[(size_t)b * 768 + i] = (unsigned short)f2bf(o);
    if (hcopy) hcopy[(size_t)b * 768 + i] = o;
  }
}

static inline void gemm(hipStream_t st,
                        const unsigned short* X1, int ldx1,
                        const unsigned short* W1, int ldw1,
                        const unsigned short* X2, int ldx2,
                        const unsigned short* W2, int ldw2, int K2,
                        const float* bias, const float* bias2, int M, int N,
                        float* outf, unsigned short* outh, int ldo, int epi,
                        const float* aux1, const float* aux2, int ldaux) {
  long tiles = (long)(M / 16) * ((N + 31) / 32);
  int grid = (int)((tiles + WPB - 1) / WPB);
  k_gemm<<<grid, BDIM, 0, st>>>(X1, ldx1, W1, ldw1, X2, ldx2, W2, ldw2, K2,
                                bias, bias2, M, N, outf, outh, ldo, epi, aux1, aux2, ldaux);
}

extern "C" void kernel_launch(void* const* d_in, const int* in_sizes, int n_in,
                              void* d_out, int out_size, void* d_ws, size_t ws_size,
                              hipStream_t stream) {
  (void)in_sizes; (void)n_in; (void)out_size; (void)ws_size;
  const float* img   = (const float*)d_in[0];
  const int*   cap   = (const int*)d_in[1];
  const float* embW  = (const float*)d_in[2];
  const float* pos   = (const float*)d_in[3];
  const float* inW   = (const float*)d_in[4];
  const float* inb   = (const float*)d_in[5];
  const float* outW  = (const float*)d_in[6];
  const float* outb  = (const float*)d_in[7];
  const float* gateW = (const float*)d_in[8];
  const float* gateb = (const float*)d_in[9];
  const float* Wih0  = (const float*)d_in[10];
  const float* Wih1  = (const float*)d_in[11];
  const float* Wih2  = (const float*)d_in[12];
  const float* Whh   = (const float*)d_in[13];
  const float* bih   = (const float*)d_in[14];
  const float* bhh   = (const float*)d_in[15];
  const float* lng   = (const float*)d_in[16];
  const float* lnb   = (const float*)d_in[17];
  const float* hwW   = (const float*)d_in[18];
  const float* hwb   = (const float*)d_in[19];
  const float* o1W   = (const float*)d_in[20];
  const float* o1b   = (const float*)d_in[21];
  const float* o2W   = (const float*)d_in[22];
  const float* o2b   = (const float*)d_in[23];

  float* logits = (float*)d_out;                           // (T,B,V)
  float* hfin   = logits + (size_t)32 * 128 * 30000;       // (T,B,H)
  float* attnw  = hfin + (size_t)32 * 128 * 768;           // (T,B,S)

  // ---- workspace carve-up (256B aligned regions) ----
  char* base = (char*)d_ws;
  size_t woff = 0;
  auto alloc = [&](size_t bytes) -> void* {
    void* p = base + woff;
    woff = (woff + bytes + 255) & ~(size_t)255;
    return p;
  };
  const size_t BE = (size_t)128 * 768;
  float*          emb    = (float*)alloc((size_t)32 * BE * 4);
  unsigned short* embbf  = (unsigned short*)alloc((size_t)32 * BE * 2);
  unsigned short* enhbf  = (unsigned short*)alloc((size_t)32 * BE * 2);
  unsigned short* hidbf  = (unsigned short*)alloc((size_t)32 * BE * 2);
  float*          h      = (float*)alloc(3 * BE * 4);
  unsigned short* hbf    = (unsigned short*)alloc(3 * BE * 2);
  float*          c      = (float*)alloc(3 * BE * 4);
  float*          q      = (float*)alloc(BE * 4);
  unsigned short* ctxpbf = (unsigned short*)alloc(BE * 2);
  float*          ctx    = (float*)alloc(BE * 4);
  unsigned short* ctxbf  = (unsigned short*)alloc(BE * 2);
  unsigned short* xbf    = (unsigned short*)alloc(BE * 2);
  float*          gates  = (float*)alloc((size_t)128 * 3072 * 4);
  float*          hraw   = (float*)alloc(BE * 4);
  const size_t SE = (size_t)128 * 196 * 768;
  unsigned short* imgbf  = (unsigned short*)alloc(SE * 2);
  unsigned short* Kb     = (unsigned short*)alloc(SE * 2);
  unsigned short* Vb     = (unsigned short*)alloc(SE * 2);
  // bf16 weights
  unsigned short* inWb   = (unsigned short*)alloc((size_t)3 * 768 * 768 * 2);
  unsigned short* outWb  = (unsigned short*)alloc((size_t)768 * 768 * 2);
  unsigned short* gateWb = (unsigned short*)alloc((size_t)768 * 1536 * 2);
  unsigned short* WihWb  = (unsigned short*)alloc((size_t)3 * 3072 * 768 * 2);
  unsigned short* WhhWb  = (unsigned short*)alloc((size_t)3 * 3072 * 768 * 2);
  unsigned short* hwWb   = (unsigned short*)alloc((size_t)768 * 1536 * 2);
  unsigned short* o1Wb   = (unsigned short*)alloc((size_t)768 * 768 * 2);
  unsigned short* o2Wb   = (unsigned short*)alloc((size_t)30000 * 768 * 2);

  auto conv = [&](const float* src, unsigned short* dst, long n) {
    long t4 = (n + 3) / 4;
    k_f2bf<<<(int)((t4 + BDIM - 1) / BDIM), BDIM, 0, stream>>>(src, dst, n);
  };
  // one-time weight + image conversions (amortized over the whole launch)
  conv(inW,   inWb,   (long)3 * 768 * 768);
  conv(outW,  outWb,  (long)768 * 768);
  conv(gateW, gateWb, (long)768 * 1536);
  conv(Wih0,  WihWb,                          (long)3072 * 768);
  conv(Wih1,  WihWb + (size_t)3072 * 768,     (long)3072 * 768);
  conv(Wih2,  WihWb + (size_t)2 * 3072 * 768, (long)3072 * 768);
  conv(Whh,   WhhWb,  (long)3 * 3072 * 768);
  conv(hwW,   hwWb,   (long)768 * 1536);
  conv(o1W,   o1Wb,   (long)768 * 768);
  conv(o2W,   o2Wb,   (long)30000 * 768);
  conv(img,   imgbf,  (long)128 * 196 * 768);

  hipMemsetAsync(h,   0, 3 * BE * sizeof(float), stream);
  hipMemsetAsync(hbf, 0, 3 * BE * sizeof(unsigned short), stream);
  hipMemsetAsync(c,   0, 3 * BE * sizeof(float), stream);

  {
    long n = (long)32 * BE;
    k_embed<<<(int)((n + BDIM - 1) / BDIM), BDIM, 0, stream>>>(cap, embW, pos, emb, embbf);
  }

  // K/V projections -> bf16 (77 MB total: resident in 192 MB L2 for the scan)
  gemm(stream, imgbf, 768, inWb + (size_t)768 * 768, 768, nullptr, 0, nullptr, 0, 0,
       inb + 768, nullptr, 25088, 768, nullptr, Kb, 768, 0, nullptr, nullptr, 0);
  gemm(stream, imgbf, 768, inWb + (size_t)2 * 768 * 768, 768, nullptr, 0, nullptr, 0, 0,
       inb + 1536, nullptr, 25088, 768, nullptr, Vb, 768, 0, nullptr, nullptr, 0);

  float*          htop   = h   + 2 * BE;
  unsigned short* htopbf = hbf + 2 * BE;
  for (int t = 0; t < 32; ++t) {
    const float*          we   = emb   + (size_t)t * BE;
    const unsigned short* webf = embbf + (size_t)t * BE;
    // q = h_top_prev @ Wq^T + bq
    gemm(stream, htopbf, 768, inWb, 768, nullptr, 0, nullptr, 0, 0,
         inb, nullptr, 128, 768, q, nullptr, 768, 0, nullptr, nullptr, 0);
    k_attn<<<128, BDIM, 0, stream>>>(q, Kb, Vb, ctxpbf, attnw + (size_t)t * 128 * 196);
    // ctx = ctx_pre @ out_W^T + out_b  (f32 + bf16)
    gemm(stream, ctxpbf, 768, outWb, 768, nullptr, 0, nullptr, 0, 0,
         outb, nullptr, 128, 768, ctx, ctxbf, 768, 0, nullptr, nullptr, 0);
    // x = g*we + (1-g)*ctx, g = sigmoid([we,ctx] @ gate_W^T + gate_b)
    gemm(stream, webf, 768, gateWb, 1536, ctxbf, 768, gateWb + 768, 1536, KDIM,
         gateb, nullptr, 128, 768, nullptr, xbf, 768, 2, we, ctx, 768);
    const unsigned short* xinbf = xbf;
    for (int l = 0; l < 3; ++l) {
      float*          hl   = h   + (size_t)l * BE;
      unsigned short* hlbf = hbf + (size_t)l * BE;
      float*          cl   = c   + (size_t)l * BE;
      // gates = x @ Wih^T + h_l @ Whh^T + bih + bhh  (two-pair K reduction)
      gemm(stream, xinbf, 768, WihWb + (size_t)l * 3072 * 768, 768,
           hlbf, 768, WhhWb + (size_t)l * 3072 * 768, 768, KDIM,
           bih + (size_t)l * 3072, bhh + (size_t)l * 3072, 128, 3072,
           gates, nullptr, 3072, 0, nullptr, nullptr, 0);
      k_lstm_point<<<(int)((128 * 768 + BDIM - 1) / BDIM), BDIM, 0, stream>>>(gates, cl, hraw);
      k_ln<<<128, BDIM, 0, stream>>>(hraw, lng + (size_t)l * 768, lnb + (size_t)l * 768,
                                     hl, hlbf, (l == 2) ? (hfin + (size_t)t * BE) : nullptr);
      xinbf = hlbf;
    }
    // enh = hg*h_top + (1-hg)*ctx, hg = sigmoid([h_top,ctx] @ hw_W^T + hw_b)
    gemm(stream, htopbf, 768, hwWb, 1536, ctxbf, 768, hwWb + 768, 1536, KDIM,
         hwb, nullptr, 128, 768, nullptr, enhbf + (size_t)t * BE, 768, 2, htop, ctx, 768);
  }

  // output MLP: GELU(enh @ o1^T + b1) @ o2^T + b2  (vocab GEMM: 189 GFLOP, dominant)
  gemm(stream, enhbf, 768, o1Wb, 768, nullptr, 0, nullptr, 0, 0,
       o1b, nullptr, 4096, 768, nullptr, hidbf, 768, 3, nullptr, nullptr, 0);
  gemm(stream, hidbf, 768, o2Wb, 768, nullptr, 0, nullptr, 0, 0,
       o2b, nullptr, 4096, 30000, logits, nullptr, 30000, 0, nullptr, nullptr, 0);
}
